// VGAE_28948079575454
// MI455X (gfx1250) — matrix operand order repaired and checked by reading, not compiled, not run on hardware
//
#include <hip/hip_runtime.h>

// ---------------- problem sizes (match reference) ----------------
#define NNODES 8192
#define NEDGES 262144
#define FEAT   512
#define NHID   256
#define NLAT   64

typedef _Float16 hf;
typedef __attribute__((ext_vector_type(16))) _Float16 v16h;
typedef __attribute__((ext_vector_type(8)))  _Float16 v8h;
typedef __attribute__((ext_vector_type(8)))  float    v8f;
typedef __attribute__((ext_vector_type(4)))  float    v4f;
typedef __attribute__((ext_vector_type(4)))  unsigned int v4u;
typedef __attribute__((ext_vector_type(4)))  unsigned int su4;   // SGPR quad (D# group 0)
typedef __attribute__((ext_vector_type(8)))  unsigned int su8;   // SGPR oct  (D# group 1)

struct U32x8 { v4u lo, hi; };   // 32B container for bit_cast -> v16h

__device__ __forceinline__ v16h frag_ld(const hf* p0, const hf* p1) {
  U32x8 t;
  t.lo = *reinterpret_cast<const v4u*>(p0);
  t.hi = *reinterpret_cast<const v4u*>(p1);
  return __builtin_bit_cast(v16h, t);
}

__device__ __forceinline__ unsigned int hash_u32(unsigned int x) {
  x ^= x >> 16; x *= 0x7feb352dU;
  x ^= x >> 15; x *= 0x846ca68bU;
  x ^= x >> 16;
  return x;
}

// LDS byte offset of a __shared__ object (flat addr low bits map to LDS offset)
__device__ __forceinline__ unsigned int lds_byte_off(const void* p) {
  return (unsigned int)(unsigned long long)p;
}

// ---------------- elementwise kernels ----------------
__global__ void __launch_bounds__(256) cast_f16_kernel(const float* __restrict__ in,
                                                       hf* __restrict__ out, int n) {
  int i = blockIdx.x * 256 + threadIdx.x;
  if (i < n) out[i] = (hf)in[i];
}

// dropout(p=0.5) + cast to f16 (scale kept values by 2)
__global__ void __launch_bounds__(256) dropout_cast_kernel(const float* __restrict__ in,
                                                           hf* __restrict__ out, int n,
                                                           unsigned int seed) {
  int i = blockIdx.x * 256 + threadIdx.x;
  if (i < n) {
    unsigned int h = hash_u32(seed ^ (unsigned int)(i * 2654435761u));
    float v = (h & 0x80000000u) ? in[i] * 2.0f : 0.0f;
    out[i] = (hf)v;
  }
}

// agg[i*C + j] = bias[j]
__global__ void __launch_bounds__(256) bias_init_kernel(float* __restrict__ agg,
                                                        const float* __restrict__ b,
                                                        int total, int C) {
  int i = blockIdx.x * 256 + threadIdx.x;
  if (i < total) agg[i] = b[i % C];
}

// edge-parallel scatter: agg[dst] += vals[e] * h[src], 4 feats / thread
// h (<=8MB) is L2-resident on a 192MB L2; atomics land in L2 atomic units.
__global__ void __launch_bounds__(256) scatter_kernel(const int* __restrict__ src,
                                                      const int* __restrict__ dst,
                                                      const float* __restrict__ vals,
                                                      const float* __restrict__ h,
                                                      float* __restrict__ agg,
                                                      int C4, int C, int total) {
  int idx = blockIdx.x * 256 + threadIdx.x;
  if (idx >= total) return;
  int e = idx / C4;
  int c = (idx % C4) * 4;
  int s = src[e], d = dst[e];
  float v = vals[e];
  v4f hv = *reinterpret_cast<const v4f*>(&h[(size_t)s * C + c]);
  float* ap = &agg[(size_t)d * C + c];
  atomicAdd(ap + 0, v * hv[0]);
  atomicAdd(ap + 1, v * hv[1]);
  atomicAdd(ap + 2, v * hv[2]);
  atomicAdd(ap + 3, v * hv[3]);
}

// z = dropout(mu + eps * exp(logvar)) -> f16
__global__ void __launch_bounds__(256) reparam_kernel(const float* __restrict__ mu,
                                                      const float* __restrict__ lv,
                                                      hf* __restrict__ zb, int n,
                                                      unsigned int seed) {
  int i = blockIdx.x * 256 + threadIdx.x;
  if (i < n) {
    unsigned int h1 = hash_u32(seed ^ (unsigned int)(i * 0x9E3779B9u));
    unsigned int h2 = hash_u32(h1 ^ 0xA511E9B3u);
    unsigned int h3 = hash_u32(h2 ^ 0x63D83595u);
    float u1 = ((h1 >> 8) + 1u) * (1.0f / 16777217.0f);
    float u2 = (h2 >> 8) * (1.0f / 16777216.0f);
    float eps = sqrtf(-2.0f * __logf(u1)) * __cosf(6.2831853f * u2);
    float z = mu[i] + eps * __expf(lv[i]);
    z = (h3 & 0x80000000u) ? z * 2.0f : 0.0f;
    zb[i] = (hf)z;
  }
}

// ---------------- WMMA GEMM: C[M,Nc] = A[M,K](f16) @ B[K,Nc](f16), f32 acc ----
// block = 256 threads = 8 waves; tile 128(M) x 64(N); K stepped by 32.
// A tile staged memory->LDS with GLOBAL_LOAD_ASYNC_TO_LDS_B128 (ASYNCcnt).
__global__ void __launch_bounds__(256) gemm_f16_wmma(const hf* __restrict__ A,
                                                     const hf* __restrict__ B,
                                                     float* __restrict__ C,
                                                     int M, int K, int Nc) {
  __shared__ __align__(16) hf As[128 * 32];   // row-major M x K-chunk
  __shared__ __align__(16) hf Bt[64 * 32];    // transposed: N x K-chunk

  const int tid  = threadIdx.x;
  const int bm   = blockIdx.y * 128;
  const int bn   = blockIdx.x * 64;
  const int lane = tid & 31;
  const int wave = tid >> 5;
  const int sel  = lane >> 4;     // half-wave select
  const int r    = lane & 15;
  const int mstrip = wave * 16;

  v8f acc[4];
  #pragma unroll
  for (int nt = 0; nt < 4; ++nt) acc[nt] = (v8f){0.f,0.f,0.f,0.f,0.f,0.f,0.f,0.f};

  for (int kb = 0; kb < K; kb += 32) {
    // stage A tile 128x32 (8KB): 512 x 16B chunks, 2 per thread, async to LDS
    #pragma unroll
    for (int i = 0; i < 2; ++i) {
      int chunk = tid * 2 + i;
      int row = chunk >> 2, c16 = chunk & 3;
      unsigned int ldso = lds_byte_off(&As[row * 32 + c16 * 8]);
      unsigned long long ga =
          (unsigned long long)&A[(size_t)(bm + row) * K + kb + c16 * 8];
      asm volatile("global_load_async_to_lds_b128 %0, %1, off"
                   :: "v"(ldso), "v"(ga) : "memory");
    }
    // stage B tile 32x64 transposed into Bt[64][32]
    {
      int k = tid >> 3, ng = tid & 7;
      v4u t = *reinterpret_cast<const v4u*>(&B[(size_t)(kb + k) * Nc + bn + ng * 8]);
      v8h th = __builtin_bit_cast(v8h, t);
      #pragma unroll
      for (int j = 0; j < 8; ++j) Bt[(ng * 8 + j) * 32 + k] = th[j];
    }
    asm volatile("s_wait_asynccnt 0" ::: "memory");
    __syncthreads();

    // A fragment: row (mstrip+r); lanes<16 hold K{0-7,16-23}, lanes>=16 K{8-15,24-31}
    v16h a = frag_ld(&As[(mstrip + r) * 32 + sel * 8],
                     &As[(mstrip + r) * 32 + sel * 8 + 16]);
    #pragma unroll
    for (int nt = 0; nt < 4; ++nt) {
      // B fragment: column (nt*16+r); lanes<16 hold K{0-7,16-23}, lanes>=16 K{8-15,24-31}
      v16h b = frag_ld(&Bt[(nt * 16 + r) * 32 + sel * 8],
                       &Bt[(nt * 16 + r) * 32 + sel * 8 + 16]);
      acc[nt] = __builtin_amdgcn_wmma_f32_16x16x32_f16(
          false, a, false, b, (short)0, acc[nt], false, false);
    }
    __syncthreads();
  }

  // C layout: VGPR v -> row v (lanes 0-15) / row v+8 (lanes 16-31), col = r
  #pragma unroll
  for (int nt = 0; nt < 4; ++nt)
    #pragma unroll
    for (int v = 0; v < 8; ++v)
      C[(size_t)(bm + mstrip + v + sel * 8) * Nc + bn + nt * 16 + r] = acc[nt][v];
}

// Build TDM descriptor group0 (128b): count=1, lds_addr, global_addr, type=2
__device__ __forceinline__ su4 tdm_group0(unsigned int lds_addr,
                                          unsigned long long gaddr) {
  su4 g;
  g[0] = 1u;                                           // count=1, user mode
  g[1] = lds_addr;                                     // lds_addr [63:32]
  g[2] = (unsigned int)(gaddr & 0xFFFFFFFFull);        // global_addr [95:64]
  g[3] = ((unsigned int)(gaddr >> 32) & 0x1FFFFFFu)    // global_addr [120:96]
         | 0x80000000u;                                // type=2 ("image") [127:126]
  return g;
}

// TDM descriptor group1 (256b): data_size=4B, 2D tensor rows x dwords,
// tile = tdim0 dwords x tdim1 rows, row stride = stride0 dwords
__device__ __forceinline__ su8 tdm_group1(unsigned int tensor_d0,
                                          unsigned int tensor_d1,
                                          unsigned int tile_d0,
                                          unsigned int tile_d1,
                                          unsigned int stride0) {
  su8 g;
  g[0] = 2u << 16;                  // data_size=2 (4 bytes); wg_mask=0; no flags
  g[1] = (tensor_d0 & 0xFFFFu) << 16;             // tensor_dim0 [79:48] low half
  g[2] = (tensor_d0 >> 16) | ((tensor_d1 & 0xFFFFu) << 16); // dim0 hi | dim1 lo
  g[3] = (tensor_d1 >> 16) | (tile_d0 << 16);     // dim1 hi | tile_dim0 [127:112]
  g[4] = tile_d1;                   // tile_dim1 [143:128]; tile_dim2=0
  g[5] = stride0;                   // tensor_dim0_stride [207:160] low 32
  g[6] = 0u;                        // stride0 hi | tensor_dim1_stride lo
  g[7] = 0u;                        // tensor_dim1_stride hi
  return g;
}

// ---------------- decode: out[N,N] = Z @ Z^T, Z is [N,64] f16 ----------------
// block = 256 threads = 8 waves; tile 128x128; K=64 resident in LDS.
// z slabs staged by the Tensor Data Mover (TENSORcnt), issued by wave 0.
__global__ void __launch_bounds__(256) decode_wmma(const hf* __restrict__ Z,
                                                   float* __restrict__ out) {
  __shared__ __align__(16) hf Az[128 * 64];
  __shared__ __align__(16) hf Bz[128 * 64];

  const int tid = threadIdx.x;
  const int m0 = blockIdx.y * 128;
  const int n0 = blockIdx.x * 128;
  const int lane = tid & 31;
  const int wave = tid >> 5;
  const int sel  = lane >> 4;
  const int r    = lane & 15;
  const int mstrip = wave * 16;

  if (wave == 0) {
    // Each z slab: 128 rows x 32 dwords, contiguous rows (stride 32 dwords)
    su8 g1 = tdm_group1(/*tensor_d0=*/32u, /*tensor_d1=*/(unsigned)NNODES,
                        /*tile_d0=*/32u, /*tile_d1=*/128u, /*stride0=*/32u);
    su4 gA = tdm_group0(lds_byte_off(Az),
                        (unsigned long long)(Z + (size_t)m0 * NLAT));
    asm volatile("tensor_load_to_lds %0, %1" :: "s"(gA), "s"(g1) : "memory");
    su4 gB = tdm_group0(lds_byte_off(Bz),
                        (unsigned long long)(Z + (size_t)n0 * NLAT));
    asm volatile("tensor_load_to_lds %0, %1" :: "s"(gB), "s"(g1) : "memory");
    __builtin_amdgcn_s_wait_tensorcnt(0);
  }
  __syncthreads();

  v8f acc[8];
  #pragma unroll
  for (int nt = 0; nt < 8; ++nt) acc[nt] = (v8f){0.f,0.f,0.f,0.f,0.f,0.f,0.f,0.f};

  #pragma unroll
  for (int kc = 0; kc < 64; kc += 32) {
    v16h a = frag_ld(&Az[(mstrip + r) * 64 + kc + sel * 8],
                     &Az[(mstrip + r) * 64 + kc + sel * 8 + 16]);
    #pragma unroll
    for (int nt = 0; nt < 8; ++nt) {
      v16h b = frag_ld(&Bz[(nt * 16 + r) * 64 + kc + sel * 8],
                       &Bz[(nt * 16 + r) * 64 + kc + sel * 8 + 16]);
      acc[nt] = __builtin_amdgcn_wmma_f32_16x16x32_f16(
          false, a, false, b, (short)0, acc[nt], false, false);
    }
  }

  #pragma unroll
  for (int nt = 0; nt < 8; ++nt)
    #pragma unroll
    for (int v = 0; v < 8; ++v)
      out[(size_t)(m0 + mstrip + v + sel * 8) * NNODES + n0 + nt * 16 + r] = acc[nt][v];
}

// ---------------- host side ----------------
extern "C" void kernel_launch(void* const* d_in, const int* in_sizes, int n_in,
                              void* d_out, int out_size, void* d_ws, size_t ws_size,
                              hipStream_t stream) {
  (void)in_sizes; (void)n_in; (void)out_size; (void)ws_size;

  const float* x    = (const float*)d_in[0];
  const int*   esrc = (const int*)d_in[1];
  const int*   edst = (const int*)d_in[2];
  const float* vals = (const float*)d_in[3];
  const float* W1   = (const float*)d_in[4];
  const float* b1   = (const float*)d_in[5];
  const float* Wmu  = (const float*)d_in[6];
  const float* bmu  = (const float*)d_in[7];
  const float* Wlv  = (const float*)d_in[8];
  const float* blv  = (const float*)d_in[9];
  float* out = (float*)d_out;

  char* ws = (char*)d_ws;
  size_t off = 0;
  auto alloc = [&](size_t bytes) -> char* {
    char* p = ws + off;
    off += (bytes + 255) & ~(size_t)255;
    return p;
  };

  hf*    xd    = (hf*)alloc(sizeof(hf) * NNODES * FEAT);
  hf*    W1h   = (hf*)alloc(sizeof(hf) * FEAT * NHID);
  hf*    Wmuh  = (hf*)alloc(sizeof(hf) * NHID * NLAT);
  hf*    Wlvh  = (hf*)alloc(sizeof(hf) * NHID * NLAT);
  float* h1    = (float*)alloc(sizeof(float) * NNODES * NHID);
  float* agg1  = (float*)alloc(sizeof(float) * NNODES * NHID);
  hf*    h1d   = (hf*)alloc(sizeof(hf) * NNODES * NHID);
  float* hmu   = (float*)alloc(sizeof(float) * NNODES * NLAT);
  float* hlv   = (float*)alloc(sizeof(float) * NNODES * NLAT);
  float* aggmu = (float*)alloc(sizeof(float) * NNODES * NLAT);
  float* agglv = (float*)alloc(sizeof(float) * NNODES * NLAT);
  hf*    zb    = (hf*)alloc(sizeof(hf) * NNODES * NLAT);

  const int T = 256;
  auto blk = [](int n) { return (n + 255) / 256; };

  // 1) dropout(x) -> f16, weights -> f16
  dropout_cast_kernel<<<blk(NNODES * FEAT), T, 0, stream>>>(x, xd, NNODES * FEAT, 0x12345u);
  cast_f16_kernel<<<blk(FEAT * NHID), T, 0, stream>>>(W1, W1h, FEAT * NHID);
  cast_f16_kernel<<<blk(NHID * NLAT), T, 0, stream>>>(Wmu, Wmuh, NHID * NLAT);
  cast_f16_kernel<<<blk(NHID * NLAT), T, 0, stream>>>(Wlv, Wlvh, NHID * NLAT);

  // 2) GCN layer 1: h1 = xd @ W1 ; agg1 = b1 + scatter(vals * h1[src] -> dst)
  gemm_f16_wmma<<<dim3(NHID / 64, NNODES / 128), T, 0, stream>>>(xd, W1h, h1,
                                                                 NNODES, FEAT, NHID);
  bias_init_kernel<<<blk(NNODES * NHID), T, 0, stream>>>(agg1, b1, NNODES * NHID, NHID);
  {
    int total = NEDGES * (NHID / 4);
    scatter_kernel<<<blk(total), T, 0, stream>>>(esrc, edst, vals, h1, agg1,
                                                 NHID / 4, NHID, total);
  }

  // 3) dropout(hid1) -> f16
  dropout_cast_kernel<<<blk(NNODES * NHID), T, 0, stream>>>(agg1, h1d,
                                                            NNODES * NHID, 0xBEEF1u);

  // 4) mu / logvar heads
  gemm_f16_wmma<<<dim3(NLAT / 64, NNODES / 128), T, 0, stream>>>(h1d, Wmuh, hmu,
                                                                 NNODES, NHID, NLAT);
  gemm_f16_wmma<<<dim3(NLAT / 64, NNODES / 128), T, 0, stream>>>(h1d, Wlvh, hlv,
                                                                 NNODES, NHID, NLAT);
  bias_init_kernel<<<blk(NNODES * NLAT), T, 0, stream>>>(aggmu, bmu, NNODES * NLAT, NLAT);
  bias_init_kernel<<<blk(NNODES * NLAT), T, 0, stream>>>(agglv, blv, NNODES * NLAT, NLAT);
  {
    int total = NEDGES * (NLAT / 4);
    scatter_kernel<<<blk(total), T, 0, stream>>>(esrc, edst, vals, hmu, aggmu,
                                                 NLAT / 4, NLAT, total);
    scatter_kernel<<<blk(total), T, 0, stream>>>(esrc, edst, vals, hlv, agglv,
                                                 NLAT / 4, NLAT, total);
  }

  // 5) reparameterize + dropout -> z (f16)
  reparam_kernel<<<blk(NNODES * NLAT), T, 0, stream>>>(aggmu, agglv, zb,
                                                       NNODES * NLAT, 0xC0FFEEu);

  // 6) decode: out = z @ z^T (write-bandwidth-bound; 128x128 tiles, WMMA + TDM)
  decode_wmma<<<dim3(NNODES / 128, NNODES / 128), T, 0, stream>>>(zb, out);
}
